// GCNEmbeddingNetwork_4750233829439
// MI455X (gfx1250) — matrix operand order repaired and checked by reading, not compiled, not run on hardware
//
#include <hip/hip_runtime.h>
#include <stdint.h>

// GCN on MI455X (gfx1250, wave32).
// N=16384, F_IN=64, HID=16, F_OUT=32; A is dense f32 but only its nonzero
// pattern matters. Pipeline:
//  k1 : read A once (1 GiB, non-temporal): per-column nonzero counts (-> deg)
//       + nonzero bitmask via v_ballot_w32, stored jw-major (bitsT[jw][i]).
//  k1b: dinv[j] = rsqrt(1 + colsum)
//  k2 : zT[f][i] = f16( dinv[i] * (x@W1)[i,f] ), transposed for the WMMA
//       B-fragment (lane-contiguous along K=i).
//  k3 : s[i] = dinv[i]*(sum_j bin[i,j]*dinv[j] + dinv[i])   (Ahat row sums)
//  k4 : H = bin^T @ Z with v_wmma_f32_16x16x32_f16; bitmask words fetched with
//       half-wave-uniform uint4 loads (one 128B line/wave/iter, L2-resident)
//       and expanded in-register to f16 {0,1} A-fragments. Epilogue applies
//       self-loop + b1 + relu and folds s[j]*h[j,k] into per-WG partials.
//  k5 : final[f] = N*b2[f] + sum_k t[k]*W2[k,f]   (layer-2 collapses since the
//       reference returns out.sum(axis=0)).
// Workspace: ~36.7 MB. All reductions fixed-order (no float atomics).

#define N_NODES 16384
#define F_INC   64
#define HIDC    16
#define F_OUTC  32
#define RCH     64                    // row chunks in pass 1
#define ROWS_PER_CH (N_NODES / RCH)   // 256
#define NJW     (N_NODES / 32)        // 512 bitmask words per row

typedef __attribute__((ext_vector_type(16))) _Float16 v16h;
typedef __attribute__((ext_vector_type(8)))  float    v8f;
typedef __attribute__((ext_vector_type(8)))  unsigned v8u;

// two bits -> packed {f16(bit_lo), f16(bit_hi)} where set bit -> 1.0h (0x3C00)
static __device__ __forceinline__ unsigned pb(unsigned wlo, unsigned whi, unsigned m) {
  return (((wlo >> m) & 1u) * 0x3C00u) | (((whi >> m) & 1u) * 0x3C000000u);
}

// ---------- k1: column-degree partials + transposed bitmask ----------
__global__ __launch_bounds__(256) void k1_deg_bits(const float* __restrict__ A,
                                                   unsigned* __restrict__ pdeg,
                                                   unsigned* __restrict__ bitsT) {
  const int j    = blockIdx.x * 256 + threadIdx.x;
  const int jw   = j >> 5;
  const int lane = threadIdx.x & 31;
  const int i0   = blockIdx.y * ROWS_PER_CH;
  unsigned cnt = 0;
  for (int i = i0; i < i0 + ROWS_PER_CH; ++i) {
    // Read-once 1 GiB stream: NT hint keeps L2 free for the bitmask we emit.
    float v = __builtin_nontemporal_load(&A[(size_t)i * N_NODES + j]);
    bool nz = (v != 0.0f);
    cnt += nz ? 1u : 0u;
    unsigned wd = __builtin_amdgcn_ballot_w32(nz); // bit b <-> column jw*32+b
    if (lane == 0) bitsT[(size_t)jw * N_NODES + i] = wd;
  }
  pdeg[(size_t)blockIdx.y * N_NODES + j] = cnt;
}

// ---------- k1b: deg -> dinv ----------
__global__ __launch_bounds__(256) void k1b_dinv(const unsigned* __restrict__ pdeg,
                                                float* __restrict__ dinv) {
  const int j = blockIdx.x * 256 + threadIdx.x;
  unsigned deg = 1u;                                // self-loop
  for (int c = 0; c < RCH; ++c) deg += pdeg[(size_t)c * N_NODES + j];
  dinv[j] = rsqrtf((float)deg);
}

// ---------- k2: zT[f][i] = f16(dinv[i] * (x @ W1)[i,f]) ----------
__global__ __launch_bounds__(256) void k2_zT(const float* __restrict__ x,
                                             const float* __restrict__ W1,
                                             const float* __restrict__ dinv,
                                             unsigned short* __restrict__ zT) {
  __shared__ float w1s[F_INC * HIDC];
  for (int t = threadIdx.x; t < F_INC * HIDC; t += 256) w1s[t] = W1[t];
  __syncthreads();
  const int i = blockIdx.x * 256 + threadIdx.x;
  float acc[HIDC];
#pragma unroll
  for (int f = 0; f < HIDC; ++f) acc[f] = 0.f;
  const float4* xp = (const float4*)(x + (size_t)i * F_INC);
#pragma unroll
  for (int k4 = 0; k4 < F_INC / 4; ++k4) {
    float4 xv = xp[k4];
#pragma unroll
    for (int f = 0; f < HIDC; ++f) {
      acc[f] += xv.x * w1s[(k4 * 4 + 0) * HIDC + f];
      acc[f] += xv.y * w1s[(k4 * 4 + 1) * HIDC + f];
      acc[f] += xv.z * w1s[(k4 * 4 + 2) * HIDC + f];
      acc[f] += xv.w * w1s[(k4 * 4 + 3) * HIDC + f];
    }
  }
  const float dv = dinv[i];
#pragma unroll
  for (int f = 0; f < HIDC; ++f) {
    _Float16 hv = (_Float16)(dv * acc[f]);
    zT[(size_t)f * N_NODES + i] = __builtin_bit_cast(unsigned short, hv);
  }
}

// ---------- k3: Ahat row sums s[i] ----------
__global__ __launch_bounds__(256) void k3_s(const unsigned* __restrict__ bitsT,
                                            const float* __restrict__ dinv,
                                            float* __restrict__ s) {
  const int i = blockIdx.x * 256 + threadIdx.x;
  float acc = 0.f;
  for (int jw = 0; jw < NJW; ++jw) {
    unsigned w = bitsT[(size_t)jw * N_NODES + i];  // coalesced across threads
    while (w) {
      int b = __builtin_ctz(w);
      w &= (w - 1u);
      acc += dinv[jw * 32 + b];                    // ~65 gathers/row, L2 hit
    }
  }
  const float di = dinv[i];
  s[i] = di * (acc + di);
}

// ---------- k4: H = bin^T @ Z via WMMA; fold s[j]*relu(...) into tpart ----------
__global__ __launch_bounds__(256) void k4_main(const unsigned* __restrict__ bitsT,
                                               const unsigned short* __restrict__ zT,
                                               const float* __restrict__ dinv,
                                               const float* __restrict__ s,
                                               const float* __restrict__ b1,
                                               float* __restrict__ tpart) {
  __shared__ float red[8][512];                    // per-wave 32x16 f32 partials
  __shared__ float cont[512];

  const int jw   = blockIdx.x;          // 32-column block of A
  const int tid  = threadIdx.x;
  const int wv   = tid >> 5;
  const int lane = tid & 31;
  const int l16  = lane & 15;           // A-frag row m / B-frag col n / D col n
  const int kg   = lane >> 4;           // half-wave selector

  v8f c0 = {};  // rows j = jw*32 + 0..15
  v8f c1 = {};  // rows j = jw*32 + 16..31

  const unsigned short* zrow = zT + (size_t)l16 * N_NODES + kg * 8;
  const unsigned* wbase = bitsT + (size_t)jw * N_NODES + kg * 8;

  for (int blk = wv; blk < NJW; blk += 8) {
    const int i0 = blk * 32;

    // B fragment: Z[i0..i0+31, 0:16]. Lane l16 = column f, K contiguous in zT.
    const uint4* zp = (const uint4*)(zrow + i0);
    uint4 q0 = zp[0], q1 = zp[2];
    v8u zb; zb[0] = q0.x; zb[1] = q0.y; zb[2] = q0.z; zb[3] = q0.w;
            zb[4] = q1.x; zb[5] = q1.y; zb[6] = q1.z; zb[7] = q1.w;
    v16h bfrag = __builtin_bit_cast(v16h, zb);

    // Bitmask words for rows i0..i0+31. Addresses are half-wave uniform:
    // lanes 0-15 fetch words {0-7,16-23}, lanes 16-31 fetch {8-15,24-31};
    // the whole wave touches one 128B line (L2-resident after k1).
    const uint4* wp = (const uint4*)(wbase + i0);
    uint4 wa0 = wp[0], wa1 = wp[1];     // K 0-7   (or 8-15)
    uint4 wb0 = wp[4], wb1 = wp[5];     // K 16-23 (or 24-31)

    // Expand bits -> f16 {0,1} A-fragments. tile0: bit l16, tile1: bit l16+16.
    const unsigned m0 = (unsigned)l16, m1 = (unsigned)l16 + 16u;
    v8u ua, ub;
    ua[0] = pb(wa0.x, wa0.y, m0); ua[1] = pb(wa0.z, wa0.w, m0);
    ua[2] = pb(wa1.x, wa1.y, m0); ua[3] = pb(wa1.z, wa1.w, m0);
    ua[4] = pb(wb0.x, wb0.y, m0); ua[5] = pb(wb0.z, wb0.w, m0);
    ua[6] = pb(wb1.x, wb1.y, m0); ua[7] = pb(wb1.z, wb1.w, m0);
    ub[0] = pb(wa0.x, wa0.y, m1); ub[1] = pb(wa0.z, wa0.w, m1);
    ub[2] = pb(wa1.x, wa1.y, m1); ub[3] = pb(wa1.z, wa1.w, m1);
    ub[4] = pb(wb0.x, wb0.y, m1); ub[5] = pb(wb0.z, wb0.w, m1);
    ub[6] = pb(wb1.x, wb1.y, m1); ub[7] = pb(wb1.z, wb1.w, m1);
    v16h a0 = __builtin_bit_cast(v16h, ua);
    v16h a1 = __builtin_bit_cast(v16h, ub);

    c0 = __builtin_amdgcn_wmma_f32_16x16x32_f16(false, a0, false, bfrag,
                                                (short)0, c0, false, false);
    c1 = __builtin_amdgcn_wmma_f32_16x16x32_f16(false, a1, false, bfrag,
                                                (short)0, c1, false, false);
  }

  // Dump accumulators: D layout -> element (m = v + 8*kg [+16 for tile1], n = l16)
#pragma unroll
  for (int v = 0; v < 8; ++v) {
    red[wv][(v + 8 * kg) * 16 + l16]      = c0[v];
    red[wv][(16 + v + 8 * kg) * 16 + l16] = c1[v];
  }
  __syncthreads();

  // Reduce 8 waves, add self-loop, bias, relu; weight by s[j].
#pragma unroll
  for (int h = 0; h < 2; ++h) {
    const int e  = tid + h * 256;
    const int mr = e >> 4, n = e & 15;
    const int j  = jw * 32 + mr;
    float sum = 0.f;
#pragma unroll
    for (int ww = 0; ww < 8; ++ww) sum += red[ww][e];
    _Float16 zh = __builtin_bit_cast(_Float16, zT[(size_t)n * N_NODES + j]);
    float hpre = sum + (float)zh;                  // + dinv[j]*y[j,n] self-loop
    float hv = fmaxf(dinv[j] * hpre + b1[n], 0.f);
    cont[e] = s[j] * hv;
  }
  __syncthreads();
  if (tid < HIDC) {
    float tsum = 0.f;
#pragma unroll
    for (int mr = 0; mr < 32; ++mr) tsum += cont[mr * 16 + tid];
    tpart[(size_t)jw * HIDC + tid] = tsum;         // deterministic, no atomics
  }
}

// ---------- k5: final[f] = N*b2[f] + sum_k t[k]*W2[k,f] ----------
__global__ __launch_bounds__(32) void k5_final(const float* __restrict__ tpart,
                                               const float* __restrict__ W2,
                                               const float* __restrict__ b2,
                                               float* __restrict__ out) {
  __shared__ float t[HIDC];
  const int tid = threadIdx.x;  // 32 threads
  if (tid < HIDC) {
    float a = 0.f;
    for (int g = 0; g < NJW; ++g) a += tpart[(size_t)g * HIDC + tid];
    t[tid] = a;
  }
  __syncthreads();
  float o = (float)N_NODES * b2[tid];
#pragma unroll
  for (int k = 0; k < HIDC; ++k) o += t[k] * W2[k * F_OUTC + tid];
  out[tid] = o;
}

extern "C" void kernel_launch(void* const* d_in, const int* in_sizes, int n_in,
                              void* d_out, int out_size, void* d_ws, size_t ws_size,
                              hipStream_t stream) {
  (void)in_sizes; (void)n_in; (void)out_size; (void)ws_size;
  const float* A  = (const float*)d_in[0];
  const float* x  = (const float*)d_in[1];
  const float* W1 = (const float*)d_in[2];
  const float* b1 = (const float*)d_in[3];
  const float* W2 = (const float*)d_in[4];
  const float* b2 = (const float*)d_in[5];
  float* out = (float*)d_out;

  // Workspace layout (all section sizes are multiples of 256B). ~36.7 MB total.
  char* ws = (char*)d_ws;
  size_t off = 0;
  unsigned*       pdeg  = (unsigned*)      (ws + off); off += (size_t)RCH * N_NODES * 4;
  float*          dinv  = (float*)         (ws + off); off += (size_t)N_NODES * 4;
  float*          svec  = (float*)         (ws + off); off += (size_t)N_NODES * 4;
  unsigned short* zT    = (unsigned short*)(ws + off); off += (size_t)HIDC * N_NODES * 2;
  float*          tpart = (float*)         (ws + off); off += (size_t)NJW * HIDC * 4;
  unsigned*       bitsT = (unsigned*)      (ws + off); off += (size_t)NJW * N_NODES * 4;

  k1_deg_bits<<<dim3(N_NODES / 256, RCH), 256, 0, stream>>>(A, pdeg, bitsT);
  k1b_dinv   <<<N_NODES / 256, 256, 0, stream>>>(pdeg, dinv);
  k2_zT      <<<N_NODES / 256, 256, 0, stream>>>(x, W1, dinv, zT);
  k3_s       <<<N_NODES / 256, 256, 0, stream>>>(bitsT, dinv, svec);
  k4_main    <<<NJW, 256, 0, stream>>>(bitsT, zT, dinv, svec, b1, tpart);
  k5_final   <<<1, 32, 0, stream>>>(tpart, W2, b2, out);
}
